// RinnaClassifierDecodeIntegrated_74749610820200
// MI455X (gfx1250) — compile-verified, matrix operand order checked by reading, not run on hardware
//
#include <hip/hip_runtime.h>
#include <hip/hip_bf16.h>
#include <math.h>

#define D   2816
#define LT  48
#define BSZ 8
#define NOUT 40
#define NPAD 48   // OUT padded to 3 WMMA n-tiles

typedef __attribute__((ext_vector_type(16))) __bf16 v16bf;
typedef __attribute__((ext_vector_type(8)))  __bf16 v8bf;
typedef __attribute__((ext_vector_type(8)))  float  v8f;
typedef __attribute__((ext_vector_type(4)))  float  f4;

// Fragment layout for 16-bit WMMA operands (ISA 7.12.2):
// lane(0..15): row/col = lane, K = kb..kb+7 (VGPR0-3) and kb+16..kb+23 (VGPR4-7)
// lane(16..31): same row/col = lane-16, kb = k0+8.
__device__ __forceinline__ v16bf load_frag(const __bf16* p) {
    union { v16bf v; v8bf h[2]; } u;
    u.h[0] = *(const v8bf*)(p);        // k = kb .. kb+7
    u.h[1] = *(const v8bf*)(p + 16);   // k = kb+16 .. kb+23
    return u.v;
}

// ---------------- kernel 1: vecs -> bf16 ----------------
__global__ void cvt_vecs(const float* __restrict__ hs, __bf16* __restrict__ Vb) {
    int idx = blockIdx.x * 256 + threadIdx.x;
    if (idx >= BSZ * LT * D) return;
    int row = idx / D, d = idx % D;
    int b = row / LT, i = row % LT;
    Vb[idx] = (__bf16)hs[((size_t)(b * (LT + 1) + i + 1)) * D + d];
}

// ---------------- kernel 2: W1 top/bot -> bf16, transposed (N-major) ----------------
__global__ void cvt_w1t(const float* __restrict__ W1,
                        __bf16* __restrict__ Ttop, __bf16* __restrict__ Tbot) {
    __shared__ float tile[32][33];
    int zoff = blockIdx.z ? D : 0;
    __bf16* T = blockIdx.z ? Tbot : Ttop;
    int k0 = blockIdx.x * 32;   // W1 row (feature/k)
    int n0 = blockIdx.y * 32;   // W1 col (hidden/n)
    for (int r = threadIdx.y; r < 32; r += 8)
        tile[r][threadIdx.x] = W1[(size_t)(zoff + k0 + r) * D + n0 + threadIdx.x];
    __syncthreads();
    for (int r = threadIdx.y; r < 32; r += 8)
        T[(size_t)(n0 + r) * D + k0 + threadIdx.x] = (__bf16)tile[threadIdx.x][r];
}

// ---------------- kernel 3: W2 -> bf16 transposed, padded to 48 rows ----------------
__global__ void cvt_w2t(const float* __restrict__ W2, __bf16* __restrict__ W2T) {
    int idx = blockIdx.x * 256 + threadIdx.x;
    if (idx >= NPAD * D) return;
    int o = idx / D, k = idx % D;
    W2T[idx] = (o < NOUT) ? (__bf16)W2[(size_t)k * NOUT + o] : (__bf16)0.0f;
}

// ---------------- kernel 4: stage-1 GEMM: Out(384xD) = Vb(384xD) x W(DxD) ----------------
// W stored N-major (row n holds k=0..D-1). blockIdx.y: 0 -> top/A, 1 -> bot/C.
// Wave tile 64x64 = 4x4 WMMA tiles. Ping-pong fragment buffers with *constant*
// buffer indices (manual 2x K-unroll) so they stay in VGPRs.
__global__ void __launch_bounds__(128)
gemm_stage1(const __bf16* __restrict__ Vb,
            const __bf16* __restrict__ Wtop, const __bf16* __restrict__ Wbot,
            float* __restrict__ Abuf, float* __restrict__ Cbuf) {
    const __bf16* W   = blockIdx.y ? Wbot : Wtop;
    float*        Out = blockIdx.y ? Cbuf : Abuf;
    int wid  = threadIdx.x >> 5;
    int lane = threadIdx.x & 31;
    int gw   = blockIdx.x * 4 + wid;          // 0..263 wave tiles
    int tm   = gw % 6;                        // 6 x 64 = 384 rows
    int tn   = gw / 6;                        // 44 x 64 = 2816 cols
    int r4    = lane & 15;
    int khalf = (lane >> 4) << 3;

    const __bf16* aptr[4];
    const __bf16* bptr[4];
#pragma unroll
    for (int x = 0; x < 4; ++x) {
        aptr[x] = Vb + (size_t)(tm * 64 + x * 16 + r4) * D + khalf;
        bptr[x] = W  + (size_t)(tn * 64 + x * 16 + r4) * D + khalf;
    }

    v8f acc[4][4] = {};
    v16bf af[2][4], bfr[2][4];
#pragma unroll
    for (int x = 0; x < 4; ++x) {             // preload k-tile 0 into buffer 0
        af[0][x]  = load_frag(aptr[x]);
        bfr[0][x] = load_frag(bptr[x]);
    }

    for (int k0 = 0; k0 < D; k0 += 64) {      // 44 iterations, 2 k-tiles each
        // issue loads for k0+32 into buffer 1
#pragma unroll
        for (int x = 0; x < 4; ++x) {
            af[1][x]  = load_frag(aptr[x] + k0 + 32);
            bfr[1][x] = load_frag(bptr[x] + k0 + 32);
            __builtin_prefetch(aptr[x] + k0 + 96, 0, 3);
            __builtin_prefetch(bptr[x] + k0 + 96, 0, 3);
        }
        // compute on buffer 0 (k0)
#pragma unroll
        for (int mi = 0; mi < 4; ++mi)
#pragma unroll
            for (int ni = 0; ni < 4; ++ni)
                acc[mi][ni] = __builtin_amdgcn_wmma_f32_16x16x32_bf16(
                    false, af[0][mi], false, bfr[0][ni], (short)0, acc[mi][ni],
                    false, false);
        // issue loads for k0+64 into buffer 0 (uniform guard)
        if (k0 + 64 < D) {
#pragma unroll
            for (int x = 0; x < 4; ++x) {
                af[0][x]  = load_frag(aptr[x] + k0 + 64);
                bfr[0][x] = load_frag(bptr[x] + k0 + 64);
            }
        }
        // compute on buffer 1 (k0+32)
#pragma unroll
        for (int mi = 0; mi < 4; ++mi)
#pragma unroll
            for (int ni = 0; ni < 4; ++ni)
                acc[mi][ni] = __builtin_amdgcn_wmma_f32_16x16x32_bf16(
                    false, af[1][mi], false, bfr[1][ni], (short)0, acc[mi][ni],
                    false, false);
    }

    // D-matrix layout: VGPR p, lanes 0-15 -> (M=p, N=lane); lanes 16-31 -> (M=p+8, N=lane-16)
    int cn = lane & 15;
    int mh = (lane >> 4) << 3;
#pragma unroll
    for (int mi = 0; mi < 4; ++mi)
#pragma unroll
        for (int ni = 0; ni < 4; ++ni)
#pragma unroll
            for (int p = 0; p < 8; ++p)
                Out[(size_t)(tm * 64 + mi * 16 + mh + p) * D + tn * 64 + ni * 16 + cn] =
                    acc[mi][ni][p];
}

// ---------------- kernel 5: fused stage-2 ----------------
// grid.x = B*L (one block per (b,i)); blockDim = 96 (3 waves, wave w owns j in [16w,16w+16))
__global__ void __launch_bounds__(96)
stage2(const float* __restrict__ Abuf, const float* __restrict__ Cbuf,
       const float* __restrict__ b1, const float* __restrict__ wlast,
       const __bf16* __restrict__ W2T, const float* __restrict__ b2,
       const int* __restrict__ spans, const int* __restrict__ avail,
       float* __restrict__ out) {
    __shared__ float pre[D];
    __shared__ float wl[D];
    __shared__ float lg[LT * NPAD];

    int b = blockIdx.x / LT, i = blockIdx.x % LT;
    const float* Arow = Abuf + (size_t)(b * LT + i) * D;
    for (int t = threadIdx.x; t < D; t += 96) {
        pre[t] = Arow[t] + b1[t];
        wl[t]  = wlast[t];
    }
    int s = spans[b * 2 + 0];
    int e = spans[b * 2 + 1];
    __syncthreads();

    int wid  = threadIdx.x >> 5;
    int lane = threadIdx.x & 31;
    int jl   = lane & 15;
    int j    = wid * 16 + jl;                 // this lane's A-fragment row
    int khalf = (lane >> 4) << 3;
    float ind = (i == s && j == e) ? 2.0f : ((s <= i && i <= j && j <= e) ? 1.0f : 0.0f);
    const float* Crow = Cbuf + (size_t)(b * LT + j) * D;
    const __bf16* w2p0 = W2T + (size_t)(0  + jl) * D + khalf;
    const __bf16* w2p1 = W2T + (size_t)(16 + jl) * D + khalf;
    const __bf16* w2p2 = W2T + (size_t)(32 + jl) * D + khalf;

    // ping-pong buffers, constant indices only
    f4    cc[2][4];
    v16bf bb[2][3];
    {
        const f4* cp = (const f4*)(Crow + khalf);
        cc[0][0] = cp[0]; cc[0][1] = cp[1]; cc[0][2] = cp[4]; cc[0][3] = cp[5];
        bb[0][0] = load_frag(w2p0);
        bb[0][1] = load_frag(w2p1);
        bb[0][2] = load_frag(w2p2);
    }

    v8f acc0 = {}, acc1 = {}, acc2 = {};
    for (int k0 = 0; k0 < D; k0 += 64) {      // 44 iterations, 2 k-tiles each
        {   // issue loads for k0+32 into buffer 1
            const f4* cp = (const f4*)(Crow + k0 + 32 + khalf);
            cc[1][0] = cp[0]; cc[1][1] = cp[1]; cc[1][2] = cp[4]; cc[1][3] = cp[5];
            bb[1][0] = load_frag(w2p0 + k0 + 32);
            bb[1][1] = load_frag(w2p1 + k0 + 32);
            bb[1][2] = load_frag(w2p2 + k0 + 32);
            __builtin_prefetch(Crow + k0 + 96 + khalf, 0, 3);
        }
        {   // compute on buffer 0 (k0)
            int kb = k0 + khalf;
            const f4* pp = (const f4*)(pre + kb);
            const f4* wp = (const f4*)(wl + kb);
            f4 p0 = pp[0], p1 = pp[1], p2 = pp[4], p3 = pp[5];
            f4 w0 = wp[0], w1 = wp[1], w2 = wp[4], w3 = wp[5];
            float hv[16];
#pragma unroll
            for (int q = 0; q < 4; ++q) {
                hv[q]      = p0[q] + cc[0][0][q] + ind * w0[q];
                hv[4 + q]  = p1[q] + cc[0][1][q] + ind * w1[q];
                hv[8 + q]  = p2[q] + cc[0][2][q] + ind * w2[q];
                hv[12 + q] = p3[q] + cc[0][3][q] + ind * w3[q];
            }
            union { v16bf v; __bf16 el[16]; } hf;
#pragma unroll
            for (int q = 0; q < 16; ++q) hf.el[q] = (__bf16)fmaxf(hv[q], 0.0f);
            acc0 = __builtin_amdgcn_wmma_f32_16x16x32_bf16(false, hf.v, false, bb[0][0],
                                                           (short)0, acc0, false, false);
            acc1 = __builtin_amdgcn_wmma_f32_16x16x32_bf16(false, hf.v, false, bb[0][1],
                                                           (short)0, acc1, false, false);
            acc2 = __builtin_amdgcn_wmma_f32_16x16x32_bf16(false, hf.v, false, bb[0][2],
                                                           (short)0, acc2, false, false);
        }
        if (k0 + 64 < D) {  // issue loads for k0+64 into buffer 0 (uniform guard)
            const f4* cp = (const f4*)(Crow + k0 + 64 + khalf);
            cc[0][0] = cp[0]; cc[0][1] = cp[1]; cc[0][2] = cp[4]; cc[0][3] = cp[5];
            bb[0][0] = load_frag(w2p0 + k0 + 64);
            bb[0][1] = load_frag(w2p1 + k0 + 64);
            bb[0][2] = load_frag(w2p2 + k0 + 64);
        }
        {   // compute on buffer 1 (k0+32)
            int kb = k0 + 32 + khalf;
            const f4* pp = (const f4*)(pre + kb);
            const f4* wp = (const f4*)(wl + kb);
            f4 p0 = pp[0], p1 = pp[1], p2 = pp[4], p3 = pp[5];
            f4 w0 = wp[0], w1 = wp[1], w2 = wp[4], w3 = wp[5];
            float hv[16];
#pragma unroll
            for (int q = 0; q < 4; ++q) {
                hv[q]      = p0[q] + cc[1][0][q] + ind * w0[q];
                hv[4 + q]  = p1[q] + cc[1][1][q] + ind * w1[q];
                hv[8 + q]  = p2[q] + cc[1][2][q] + ind * w2[q];
                hv[12 + q] = p3[q] + cc[1][3][q] + ind * w3[q];
            }
            union { v16bf v; __bf16 el[16]; } hf;
#pragma unroll
            for (int q = 0; q < 16; ++q) hf.el[q] = (__bf16)fmaxf(hv[q], 0.0f);
            acc0 = __builtin_amdgcn_wmma_f32_16x16x32_bf16(false, hf.v, false, bb[1][0],
                                                           (short)0, acc0, false, false);
            acc1 = __builtin_amdgcn_wmma_f32_16x16x32_bf16(false, hf.v, false, bb[1][1],
                                                           (short)0, acc1, false, false);
            acc2 = __builtin_amdgcn_wmma_f32_16x16x32_bf16(false, hf.v, false, bb[1][2],
                                                           (short)0, acc2, false, false);
        }
    }

    // spill logits to LDS (D-matrix layout)
    int n15 = lane & 15;
    int mh  = (lane >> 4) << 3;
#pragma unroll
    for (int p = 0; p < 8; ++p) {
        int jj = wid * 16 + mh + p;
        lg[jj * NPAD +  0 + n15] = acc0[p];
        lg[jj * NPAD + 16 + n15] = acc1[p];
        lg[jj * NPAD + 32 + n15] = acc2[p];
    }
    __syncthreads();

    // log-softmax + mask, one thread per j row
    if (threadIdx.x < LT) {
        int jr = threadIdx.x;
        float v[NOUT];
        float m = -INFINITY;
#pragma unroll
        for (int o = 0; o < NOUT; ++o) {
            v[o] = lg[jr * NPAD + o] + b2[o];
            m = fmaxf(m, v[o]);
        }
        float sum = 0.0f;
#pragma unroll
        for (int o = 0; o < NOUT; ++o) sum += __expf(v[o] - m);
        float lse = m + __logf(sum);
        bool msk = avail[i * LT + jr] >= 1;
        float* op = out + ((size_t)(b * LT + i) * LT + jr) * NOUT;
#pragma unroll
        for (int o = 0; o < NOUT; ++o) op[o] = msk ? (v[o] - lse) : -1.0f;
    }
}

extern "C" void kernel_launch(void* const* d_in, const int* in_sizes, int n_in,
                              void* d_out, int out_size, void* d_ws, size_t ws_size,
                              hipStream_t stream) {
    const float* hs    = (const float*)d_in[0];   // (8,49,2816)
    const float* W1    = (const float*)d_in[1];   // (5633,2816)
    const float* b1    = (const float*)d_in[2];   // (2816)
    const float* W2    = (const float*)d_in[3];   // (2816,40)
    const float* b2    = (const float*)d_in[4];   // (40)
    const int*   spans = (const int*)d_in[5];     // (8,2)
    const int*   avail = (const int*)d_in[6];     // (48,48)
    float* out = (float*)d_out;                   // (8,48,48,40)

    char* ws = (char*)d_ws;
    const size_t szVb = (size_t)BSZ * LT * D * 2;       // 2,162,688
    const size_t szWT = (size_t)D * D * 2;              // 15,859,712
    const size_t szW2 = (size_t)NPAD * D * 2;           // 270,336
    const size_t szAC = (size_t)BSZ * LT * D * 4;       // 4,325,376
    __bf16* Vb    = (__bf16*)(ws);
    __bf16* W1top = (__bf16*)(ws + szVb);
    __bf16* W1bot = (__bf16*)(ws + szVb + szWT);
    __bf16* W2T   = (__bf16*)(ws + szVb + 2 * szWT);
    float*  Abuf  = (float*) (ws + szVb + 2 * szWT + szW2);
    float*  Cbuf  = (float*) (ws + szVb + 2 * szWT + szW2 + szAC);
    const float* wlast = W1 + (size_t)(2 * D) * D;      // row 5632 of W1 (fp32)

    {   // V -> bf16
        int n = BSZ * LT * D;
        cvt_vecs<<<(n + 255) / 256, 256, 0, stream>>>(hs, Vb);
    }
    {   // W1 top/bot -> bf16 transposed
        dim3 g(D / 32, D / 32, 2), blk(32, 8, 1);
        cvt_w1t<<<g, blk, 0, stream>>>(W1, W1top, W1bot);
    }
    {   // W2 -> bf16 transposed + padded
        int n = NPAD * D;
        cvt_w2t<<<(n + 255) / 256, 256, 0, stream>>>(W2, W2T);
    }
    {   // stage-1 GEMMs (A and C)
        dim3 g(66, 2, 1);
        gemm_stage1<<<g, 128, 0, stream>>>(Vb, W1top, W1bot, Abuf, Cbuf);
    }
    {   // fused stage-2
        stage2<<<BSZ * LT, 96, 0, stream>>>(Abuf, Cbuf, b1, wlast, W2T, b2,
                                            spans, avail, out);
    }
}
